// myNNConv_2327872274901
// MI455X (gfx1250) — matrix operand order repaired and checked by reading, not compiled, not run on hardware
//
#include <hip/hip_runtime.h>
#include <hip/hip_bf16.h>
#include <math.h>

#define N_NODES 50000
#define N_EDGES 800000
#define IN_C    128
#define OUT_C   128
#define K1      256           // 2*IN_C
#define EB      64            // edges per block
#define BN_EPS  1e-5f
#define STAT_BLOCKS 200
#define ROWS_PER_STAT 250     // 200*250 = 50000

typedef __attribute__((ext_vector_type(16))) __bf16 v16bf;
typedef __attribute__((ext_vector_type(8)))  float  v8f;

union Frag {
    v16bf v;
    uint4 q[2];
};

__device__ __forceinline__ unsigned short f32_to_bf16_bits(float f) {
    unsigned u = __float_as_uint(f);
    u += 0x7FFFu + ((u >> 16) & 1u);   // round-to-nearest-even
    return (unsigned short)(u >> 16);
}

// ---------------------------------------------------------------------------
// Weight prep: bf16, transposed to N-major so B-fragments are contiguous.
// w1t[n][k] (128x256), w2t[n][k] (128x128)
// ---------------------------------------------------------------------------
__global__ void prep_weights(const float* __restrict__ W1,
                             const float* __restrict__ W2,
                             unsigned short* __restrict__ w1t,
                             unsigned short* __restrict__ w2t) {
    int i = blockIdx.x * blockDim.x + threadIdx.x;
    if (i < K1 * OUT_C) {                 // 32768
        int n = i / K1, k = i % K1;
        w1t[i] = f32_to_bf16_bits(W1[k * OUT_C + n]);
    }
    if (i < IN_C * OUT_C) {               // 16384
        int n = i / IN_C, k = i % IN_C;
        w2t[i] = f32_to_bf16_bits(W2[k * OUT_C + n]);
    }
}

__global__ void init_out(float* __restrict__ out) {
    size_t i = (size_t)blockIdx.x * blockDim.x + threadIdx.x;
    if (i < (size_t)N_NODES * OUT_C) out[i] = -INFINITY;
}

// ---------------------------------------------------------------------------
// Fused EdgeConv kernel: feature build -> GEMM1 -> Mish -> GEMM2 -> scatter-max
// ---------------------------------------------------------------------------
__launch_bounds__(256, 1)
__global__ void edge_kernel(const float* __restrict__ x,
                            const int* __restrict__ eidx,
                            const unsigned short* __restrict__ w1t,
                            const unsigned short* __restrict__ w2t,
                            const float* __restrict__ b1,
                            const float* __restrict__ b2,
                            float* __restrict__ out) {
    __shared__ unsigned short featS[EB][K1 + 8];    // 64 x 264 bf16 bits
    __shared__ unsigned short hS[EB][OUT_C + 8];    // 64 x 136 bf16 bits
    __shared__ int dstS[EB];

    const int tid = threadIdx.x;
    const int e0  = blockIdx.x * EB;

    // ---- gather x_i / x_j, build bf16 feature tile [x_i, x_j - x_i] ----
    {
        int el = tid >> 2;            // edge within tile (4 threads/edge)
        int q  = tid & 3;
        int c0 = q * 32;
        int src = eidx[e0 + el];
        int dst = eidx[N_EDGES + e0 + el];
        if (q == 0) dstS[el] = dst;
        const float* xi = x + (size_t)dst * IN_C;
        const float* xj = x + (size_t)src * IN_C;
#pragma unroll
        for (int c = 0; c < 32; c += 4) {
            float4 a = *(const float4*)(xi + c0 + c);
            float4 b = *(const float4*)(xj + c0 + c);
            featS[el][c0 + c + 0] = f32_to_bf16_bits(a.x);
            featS[el][c0 + c + 1] = f32_to_bf16_bits(a.y);
            featS[el][c0 + c + 2] = f32_to_bf16_bits(a.z);
            featS[el][c0 + c + 3] = f32_to_bf16_bits(a.w);
            featS[el][IN_C + c0 + c + 0] = f32_to_bf16_bits(b.x - a.x);
            featS[el][IN_C + c0 + c + 1] = f32_to_bf16_bits(b.y - a.y);
            featS[el][IN_C + c0 + c + 2] = f32_to_bf16_bits(b.z - a.z);
            featS[el][IN_C + c0 + c + 3] = f32_to_bf16_bits(b.w - a.w);
        }
    }
    __syncthreads();

    const int wave  = tid >> 5;
    const int lane  = tid & 31;
    const int n0    = wave * 16;      // this wave's 16-column slab
    const int col16 = lane & 15;
    const int hi    = lane >> 4;      // lane half selects K sub-group

    const v8f vzero = {0.f, 0.f, 0.f, 0.f, 0.f, 0.f, 0.f, 0.f};

    // ---------------- GEMM1: feat(64x256) @ W1(256x128) ----------------
    v8f acc[4] = {vzero, vzero, vzero, vzero};
    const unsigned short* w1col = w1t + (size_t)(n0 + col16) * K1;
#pragma unroll
    for (int kk = 0; kk < 8; ++kk) {
        Frag bf;
        const int kb = kk * 32 + hi * 8;           // B: K 0-7 then 16-23 (shifted by 8 for hi lanes)
        bf.q[0] = *(const uint4*)(w1col + kb);
        bf.q[1] = *(const uint4*)(w1col + kb + 16);
#pragma unroll
        for (int r = 0; r < 4; ++r) {
            Frag af;
            const int row = r * 16 + col16;
            const int ka  = kk * 32 + hi * 8;      // A: K 0-7 then 16-23 (shifted by 8 for hi lanes)
            af.q[0] = *(const uint4*)&featS[row][ka];
            af.q[1] = *(const uint4*)&featS[row][ka + 16];
            acc[r] = __builtin_amdgcn_wmma_f32_16x16x32_bf16(
                false, af.v, false, bf.v, (short)0, acc[r], false, false);
        }
    }

    // ---- bias + Mish, write bf16 h tile ----
    const float bias1 = b1[n0 + col16];
#pragma unroll
    for (int r = 0; r < 4; ++r) {
#pragma unroll
        for (int vi = 0; vi < 8; ++vi) {
            float v  = acc[r][vi] + bias1;
            float sp = __logf(1.0f + __expf(v));   // softplus
            float mv = v * tanhf(sp);              // mish
            int row  = r * 16 + hi * 8 + vi;       // C/D layout: M = vi + 8*hi
            hS[row][n0 + col16] = f32_to_bf16_bits(mv);
        }
    }
    __syncthreads();

    // ---------------- GEMM2: h(64x128) @ W2(128x128) ----------------
    v8f acc2[4] = {vzero, vzero, vzero, vzero};
    const unsigned short* w2col = w2t + (size_t)(n0 + col16) * OUT_C;
#pragma unroll
    for (int kk = 0; kk < 4; ++kk) {
        Frag bf;
        const int kb = kk * 32 + hi * 8;
        bf.q[0] = *(const uint4*)(w2col + kb);
        bf.q[1] = *(const uint4*)(w2col + kb + 16);
#pragma unroll
        for (int r = 0; r < 4; ++r) {
            Frag af;
            const int row = r * 16 + col16;
            const int ka  = kk * 32 + hi * 8;
            af.q[0] = *(const uint4*)&hS[row][ka];
            af.q[1] = *(const uint4*)&hS[row][ka + 16];
            acc2[r] = __builtin_amdgcn_wmma_f32_16x16x32_bf16(
                false, af.v, false, bf.v, (short)0, acc2[r], false, false);
        }
    }

    // ---- bias + deterministic scatter-max (order-preserving int atomics) ----
    const float bias2 = b2[n0 + col16];
#pragma unroll
    for (int r = 0; r < 4; ++r) {
#pragma unroll
        for (int vi = 0; vi < 8; ++vi) {
            float v  = acc2[r][vi] + bias2;
            int   el = r * 16 + hi * 8 + vi;
            int   dst = dstS[el];
            float* addr = out + (size_t)dst * OUT_C + n0 + col16;
            int iv = __float_as_int(v);
            if (iv >= 0) atomicMax((int*)addr, iv);
            else         atomicMin((unsigned int*)addr, __float_as_uint(v));
        }
    }
}

// ---------------------------------------------------------------------------
// BatchNorm: deterministic two-stage column stats, then normalize in place.
// ---------------------------------------------------------------------------
__global__ void stats_partial(const float* __restrict__ agg,
                              float* __restrict__ psum,
                              float* __restrict__ psumsq) {
    __shared__ float s1[256];
    __shared__ float s2[256];
    int tid  = threadIdx.x;
    int col  = tid & 127;
    int half = tid >> 7;
    int r0   = blockIdx.x * ROWS_PER_STAT;
    float s = 0.f, q = 0.f;
    for (int r = r0 + half; r < r0 + ROWS_PER_STAT; r += 2) {
        float v = agg[(size_t)r * OUT_C + col];
        if (v == -INFINITY) v = 0.f;           // isolated node -> 0
        s += v;
        q += v * v;
    }
    s1[tid] = s; s2[tid] = q;
    __syncthreads();
    if (half == 0) {
        psum[blockIdx.x * OUT_C + col]   = s1[col] + s1[col + 128];
        psumsq[blockIdx.x * OUT_C + col] = s2[col] + s2[col + 128];
    }
}

__global__ void stats_final(const float* __restrict__ psum,
                            const float* __restrict__ psumsq,
                            float* __restrict__ meanv,
                            float* __restrict__ invstdv) {
    int col = threadIdx.x;                     // 128 threads
    float s = 0.f, q = 0.f;
    for (int b = 0; b < STAT_BLOCKS; ++b) {
        s += psum[b * OUT_C + col];
        q += psumsq[b * OUT_C + col];
    }
    float mean = s / (float)N_NODES;
    float var  = q / (float)N_NODES - mean * mean;   // biased
    if (var < 0.f) var = 0.f;
    meanv[col]   = mean;
    invstdv[col] = rsqrtf(var + BN_EPS);
}

__global__ void normalize_kernel(float* __restrict__ out,
                                 const float* __restrict__ meanv,
                                 const float* __restrict__ invstdv,
                                 const float* __restrict__ gamma,
                                 const float* __restrict__ beta) {
    size_t i = (size_t)blockIdx.x * blockDim.x + threadIdx.x;
    if (i >= (size_t)N_NODES * OUT_C) return;
    int col = (int)(i & 127);
    float v = out[i];
    if (v == -INFINITY) v = 0.f;
    out[i] = (v - meanv[col]) * invstdv[col] * gamma[col] + beta[col];
}

// ---------------------------------------------------------------------------
extern "C" void kernel_launch(void* const* d_in, const int* in_sizes, int n_in,
                              void* d_out, int out_size, void* d_ws, size_t ws_size,
                              hipStream_t stream) {
    const float* x     = (const float*)d_in[0];
    const int*   eidx  = (const int*)d_in[1];
    const float* W1    = (const float*)d_in[2];
    const float* b1    = (const float*)d_in[3];
    const float* W2    = (const float*)d_in[4];
    const float* b2    = (const float*)d_in[5];
    const float* gamma = (const float*)d_in[6];
    const float* beta  = (const float*)d_in[7];
    float* out = (float*)d_out;

    // workspace layout
    char* ws = (char*)d_ws;
    unsigned short* w1t = (unsigned short*)(ws);                 // 65536 B
    unsigned short* w2t = (unsigned short*)(ws + 65536);         // 32768 B
    float* psum    = (float*)(ws + 98304);                       // 200*128*4
    float* psumsq  = (float*)(ws + 98304 + STAT_BLOCKS*OUT_C*4);
    float* meanv   = (float*)(ws + 98304 + 2*STAT_BLOCKS*OUT_C*4);
    float* invstdv = meanv + OUT_C;

    prep_weights<<<(K1*OUT_C + 255)/256, 256, 0, stream>>>(W1, W2, w1t, w2t);

    init_out<<<((size_t)N_NODES*OUT_C + 255)/256, 256, 0, stream>>>(out);

    edge_kernel<<<N_EDGES / EB, 256, 0, stream>>>(x, eidx, w1t, w2t, b1, b2, out);

    stats_partial<<<STAT_BLOCKS, 256, 0, stream>>>(out, psum, psumsq);
    stats_final<<<1, 128, 0, stream>>>(psum, psumsq, meanv, invstdv);
    normalize_kernel<<<((size_t)N_NODES*OUT_C + 255)/256, 256, 0, stream>>>(
        out, meanv, invstdv, gamma, beta);
}
